// SparseSemanticFieldCore_69982197121714
// MI455X (gfx1250) — matrix operand order repaired and checked by reading, not compiled, not run on hardware
//
#include <hip/hip_runtime.h>

#define M_ 512
#define C_ 256
#define B_ 8
#define T_ 32
#define MC_ (M_*C_)      /* 131072 */
#define TOPK_ 8
#define LAM_ 0.05f
#define RHO_ 0.9f
#define DLR_ 0.03f
#define EPS_ 1e-12f
#define LNEPS_ 1e-5f

typedef __attribute__((ext_vector_type(16))) __bf16 v16bf;
typedef __attribute__((ext_vector_type(8)))  float  v8f;

union BF16Frag { uint4 q[2]; unsigned short s[16]; v16bf v; };

__device__ __forceinline__ unsigned short f2bf(float f) {
  unsigned int x = __float_as_uint(f);
  x += 0x7fffu + ((x >> 16) & 1u);          // round-to-nearest-even
  return (unsigned short)(x >> 16);
}

// ---------------- f32 -> bf16 bulk convert (8 elems/thread) ----------------
__global__ __launch_bounds__(256) void k_f32_to_bf16(const float* __restrict__ src,
                                                     unsigned short* __restrict__ dst,
                                                     size_t n8) {
  size_t i = (size_t)blockIdx.x * 256 + threadIdx.x;
  if (i >= n8) return;
  const float4* s4 = (const float4*)src;
  float4 a = s4[2*i], b = s4[2*i+1];
  uint4 o;
  o.x = (unsigned)f2bf(a.x) | ((unsigned)f2bf(a.y) << 16);
  o.y = (unsigned)f2bf(a.z) | ((unsigned)f2bf(a.w) << 16);
  o.z = (unsigned)f2bf(b.x) | ((unsigned)f2bf(b.y) << 16);
  o.w = (unsigned)f2bf(b.z) | ((unsigned)f2bf(b.w) << 16);
  ((uint4*)dst)[i] = o;
}

// ---------------- D0 = column-normalized base dictionary, broadcast over B --
__global__ __launch_bounds__(128) void k_init_D(const float* __restrict__ base,
                                                float* __restrict__ D) {
  int c = blockIdx.x, tid = threadIdx.x;
  __shared__ float red[128];
  float s = 0.f;
  for (int m = tid; m < M_; m += 128) { float v = base[m*C_+c]; s += v*v; }
  red[tid] = s; __syncthreads();
  for (int o = 64; o; o >>= 1) { if (tid < o) red[tid] += red[tid+o]; __syncthreads(); }
  float inv = 1.f / fmaxf(sqrtf(red[0]), EPS_);
  for (int m = tid; m < M_; m += 128) {
    float v = base[m*C_+c] * inv;
    for (int b = 0; b < B_; b++) D[((size_t)(b*M_+m))*C_ + c] = v;
  }
}

// ---------------- P[bt][j] = u_bt @ W_cand_u.T + b_cand_u + b_cand_c -------
// One wave per 16x16 output tile, K=512 in 16 WMMA steps (bf16 -> f32 acc).
__global__ __launch_bounds__(128) void k_precompute_P(
    const unsigned short* __restrict__ ubf, const unsigned short* __restrict__ wubf,
    const float* __restrict__ bcu, const float* __restrict__ bcc,
    float* __restrict__ P) {
  int lane = threadIdx.x & 31, wv = threadIdx.x >> 5;
  int jt = blockIdx.x * 4 + wv;          // 0..8191
  int rbase = blockIdx.y * 16;           // row tile over bt (0..255)
  int jbase = jt * 16;
  int rr = lane & 15, hi = lane >> 4;
  const unsigned short* arow = ubf  + (size_t)(rbase + rr) * M_;
  const unsigned short* brow = wubf + (size_t)(jbase + rr) * M_;
  v8f acc = {0,0,0,0,0,0,0,0};
  for (int kk = 0; kk < 16; ++kk) {
    int k0 = kk * 32;
    BF16Frag A, Bf;
    int ab = k0 + (hi ? 8 : 0);          // A: lanes<16 -> K 0..7,16..23 ; else 8..15,24..31
    A.q[0] = *(const uint4*)(arow + ab);
    A.q[1] = *(const uint4*)(arow + ab + 16);
    int bb = k0 + (hi ? 8 : 0);          // B: lanes<16 -> K 0..7,16..23 ; else 8..15,24..31
    Bf.q[0] = *(const uint4*)(brow + bb);
    Bf.q[1] = *(const uint4*)(brow + bb + 16);
    acc = __builtin_amdgcn_wmma_f32_16x16x32_bf16(false, A.v, false, Bf.v,
                                                  (short)0, acc, false, false);
  }
  int j = jbase + rr;
  float bias = bcu[j] + bcc[j];
  int Rw = rbase + (hi ? 8 : 0);
  for (int v = 0; v < 8; ++v)
    P[(size_t)(Rw + v) * MC_ + j] = acc[v] + bias;
}

// ---------------- step kernel 1: LN, logits, shrink, top-8, a_t, c_t -------
__global__ __launch_bounds__(256) void k_step1(
    const float* __restrict__ u, const float* __restrict__ valid,
    const float* __restrict__ Wcu, const float* __restrict__ Wcc,
    const float* __restrict__ lnug, const float* __restrict__ lnub,
    const float* __restrict__ lncg, const float* __restrict__ lncb,
    float* __restrict__ cbuf, float* __restrict__ abuf,
    int* __restrict__ tki, float* __restrict__ tkv,
    float* __restrict__ css, float* __restrict__ dss, float* __restrict__ xss,
    float* __restrict__ errsq, int t) {
  __shared__ float s_lnu[B_][M_];
  __shared__ float s_lnc[B_][C_];
  __shared__ float s_s[B_][C_];
  __shared__ float s_abs[B_][C_];
  int tid = threadIdx.x;
  for (int i = tid; i < B_*C_; i += 256) { css[i]=0.f; dss[i]=0.f; xss[i]=0.f; }
  if (tid < B_) errsq[tid] = 0.f;
  int b = tid >> 5, lane = tid & 31;      // one wave32 per batch row
  const float* urow = u + ((size_t)(b*T_+t))*M_;
  float uv[16]; float sum = 0.f;
  for (int i = 0; i < 16; i++) { uv[i] = urow[lane + 32*i]; sum += uv[i]; }
  for (int o = 16; o; o >>= 1) sum += __shfl_xor(sum, o);
  float mu = sum * (1.f/M_), var = 0.f;
  for (int i = 0; i < 16; i++) { float d = uv[i]-mu; var += d*d; }
  for (int o = 16; o; o >>= 1) var += __shfl_xor(var, o);
  float rstd = rsqrtf(var*(1.f/M_) + LNEPS_);
  for (int i = 0; i < 16; i++) { int m = lane + 32*i;
    s_lnu[b][m] = (uv[i]-mu)*rstd*lnug[m] + lnub[m]; }
  float cp[8]; float sc = 0.f;
  for (int i = 0; i < 8; i++) { cp[i] = cbuf[b*C_ + lane + 32*i]; sc += cp[i]; }
  for (int o = 16; o; o >>= 1) sc += __shfl_xor(sc, o);
  float muc = sc*(1.f/C_), vc = 0.f;
  for (int i = 0; i < 8; i++) { float d = cp[i]-muc; vc += d*d; }
  for (int o = 16; o; o >>= 1) vc += __shfl_xor(vc, o);
  float rc = rsqrtf(vc*(1.f/C_) + LNEPS_);
  for (int i = 0; i < 8; i++) { int c = lane + 32*i;
    s_lnc[b][c] = (cp[i]-muc)*rc*lncg[c] + lncb[c]; }
  __syncthreads();
  for (int i = 0; i < 8; i++) {
    int c = lane + 32*i;
    float lg = 0.f;
    const float* wr = Wcu + (size_t)c*M_;
    for (int m = 0; m < M_; m++) lg += s_lnu[b][m]*wr[m];
    const float* wc = Wcc + (size_t)c*C_;
    for (int k = 0; k < C_; k++) lg += s_lnc[b][k]*wc[k];
    float ab = fabsf(lg) - LAM_;
    float sh = ab > 0.f ? copysignf(ab, lg) : 0.f;
    s_s[b][c] = sh; s_abs[b][c] = fabsf(sh);
  }
  for (int it = 0; it < TOPK_; ++it) {     // iterative argmax top-k
    float mv = -1.f; int mi = 0;
    for (int i = 0; i < 8; i++) {
      int c = lane + 32*i; float v = s_abs[b][c];
      if (v > mv || (v == mv && c < mi)) { mv = v; mi = c; }
    }
    for (int o = 16; o; o >>= 1) {
      float ov = __shfl_xor(mv, o); int oi = __shfl_xor(mi, o);
      if (ov > mv || (ov == mv && oi < mi)) { mv = ov; mi = oi; }
    }
    if (lane == (mi & 31)) s_abs[b][mi] = -2.f;     // sentinel = selected
    if (lane == 0) { tki[b*TOPK_+it] = mi; tkv[b*TOPK_+it] = s_s[b][mi]; }
  }
  float v = valid[b*T_ + t];
  for (int i = 0; i < 8; i++) {
    int c = lane + 32*i;
    float a = (s_abs[b][c] == -2.f) ? s_s[b][c] : 0.f;
    abuf[b*C_+c] = a;
    float cn = RHO_*cp[i] + (1.f-RHO_)*a;
    cbuf[b*C_+c] = v*cn + (1.f-v)*cp[i];
  }
}

// ---------------- step kernel 2: u_hat (8-sparse), r, err^2 ----------------
__global__ __launch_bounds__(256) void k_step2(
    const float* __restrict__ u, const float* __restrict__ D,
    const int* __restrict__ tki, const float* __restrict__ tkv,
    float* __restrict__ rbuf, float* __restrict__ errsq, int t) {
  int idx = blockIdx.x*256 + threadIdx.x;   // 0..B*M-1
  int b = idx >> 9, m = idx & (M_-1);
  const float* Drow = D + ((size_t)(b*M_+m))*C_;
  float uh = 0.f;
  for (int k = 0; k < TOPK_; k++) uh += Drow[tki[b*TOPK_+k]] * tkv[b*TOPK_+k];
  float r = u[((size_t)(b*T_+t))*M_ + m] - uh;
  rbuf[b*M_+m] = r;
  atomicAdd(&errsq[b], r*r);
}

// ---------------- step kernel 3: gate ---------------------------------------
__global__ __launch_bounds__(256) void k_step3(
    const float* __restrict__ u, const float* __restrict__ cbuf,
    const float* __restrict__ Wg, const float* __restrict__ bg,
    const float* __restrict__ errsq, float* __restrict__ gbuf, int t) {
  int b = threadIdx.x >> 5, lane = threadIdx.x & 31;
  float p = 0.f;
  for (int i = 0; i < 24; i++) {
    int d = lane + 32*i;                    // 0..767
    float x = (d < M_) ? u[((size_t)(b*T_+t))*M_ + d] : cbuf[b*C_ + (d - M_)];
    p += x * Wg[d];
  }
  for (int o = 16; o; o >>= 1) p += __shfl_xor(p, o);
  if (lane == 0) {
    float z = p + sqrtf(errsq[b]) * Wg[M_+C_] + bg[0];
    gbuf[b] = 1.f / (1.f + expf(-z));
  }
}

// ---------------- step kernel 4a: cand GEMM (bf16 WMMA, W_cand_c L2-resident)
// grid (16 c-blocks, 16 m-chunks), 4 waves/block; accumulates per-(b,c)
// Sum(cand^2), Sum(dl^2), Sum(dl*cand) so the blend norm is analytic.
__global__ __launch_bounds__(128) void k_step4a(
    const unsigned short* __restrict__ wcbf, const float* __restrict__ P,
    const float* __restrict__ cbuf, const float* __restrict__ abuf,
    const float* __restrict__ D, const float* __restrict__ rbuf,
    float* __restrict__ CAND,
    float* __restrict__ css, float* __restrict__ dss, float* __restrict__ xss,
    int t) {
  __shared__ float s_css[16][B_], s_dss[16][B_], s_x[16][B_];
  int tid = threadIdx.x, lane = tid & 31, wv = tid >> 5;
  int c0 = blockIdx.x * 16, mchunk = blockIdx.y;
  int rr = lane & 15, hi = lane >> 4;
  { int cI = tid >> 3, bI = tid & 7; s_css[cI][bI]=0.f; s_dss[cI][bI]=0.f; s_x[cI][bI]=0.f; }
  __syncthreads();
  BF16Frag af[8];                           // A = padded c_t (16x256 bf16), hoisted
  for (int kk = 0; kk < 8; kk++) {
    int k0 = kk*32 + (hi ? 8 : 0);
    if (rr < B_) {
      const float* crow = cbuf + rr*C_;
      for (int e = 0; e < 8; e++) af[kk].s[e]   = f2bf(crow[k0 + e]);
      for (int e = 0; e < 8; e++) af[kk].s[8+e] = f2bf(crow[k0 + 16 + e]);
    } else {
      for (int e = 0; e < 16; e++) af[kk].s[e] = 0;
    }
  }
  float l_css[8], l_dss[8], l_x[8];
  for (int v = 0; v < 8; v++) { l_css[v]=0.f; l_dss[v]=0.f; l_x[v]=0.f; }
  int c = c0 + rr;
  for (int mi = 0; mi < 8; ++mi) {
    int m = mchunk*32 + wv*8 + mi;
    int j = m*C_ + c0 + rr;                 // row of W_cand_c for lane's column
    const unsigned short* brow = wcbf + (size_t)j*C_ + (hi ? 8 : 0);
    __builtin_prefetch((const void*)(brow + (size_t)C_*C_), 0, 1); // next m row
    v8f acc = {0,0,0,0,0,0,0,0};
    for (int kk = 0; kk < 8; kk++) {
      BF16Frag Bf;
      Bf.q[0] = *(const uint4*)(brow + kk*32);
      Bf.q[1] = *(const uint4*)(brow + kk*32 + 16);
      acc = __builtin_amdgcn_wmma_f32_16x16x32_bf16(false, af[kk].v, false, Bf.v,
                                                    (short)0, acc, false, false);
    }
    if (!hi) {                              // lanes 0-15: rows 0-7 = real batches
      for (int v = 0; v < 8; v++) {
        int bb = v;
        float cand = acc[v] + P[((size_t)(bb*T_+t))*MC_ + (size_t)m*C_ + c];
        size_t di = ((size_t)(bb*M_+m))*C_ + c;
        CAND[di] = cand;
        float dl = D[di] + DLR_ * rbuf[bb*M_+m] * abuf[bb*C_+c];
        l_css[v] += cand*cand; l_dss[v] += dl*dl; l_x[v] += dl*cand;
      }
    }
  }
  if (!hi) {
    for (int v = 0; v < 8; v++) {
      atomicAdd(&s_css[rr][v], l_css[v]);   // LDS ds_add_f32
      atomicAdd(&s_dss[rr][v], l_dss[v]);
      atomicAdd(&s_x[rr][v],   l_x[v]);
    }
  }
  __syncthreads();
  { int cI = tid >> 3, bI = tid & 7;        // 128 threads = 16c x 8b exactly
    atomicAdd(&css[bI*C_ + c0 + cI], s_css[cI][bI]);
    atomicAdd(&dss[bI*C_ + c0 + cI], s_dss[cI][bI]);
    atomicAdd(&xss[bI*C_ + c0 + cI], s_x[cI][bI]); }
}

// ---------------- step kernel 4b: normalize/blend/update D, z = D_next . c_t
__global__ __launch_bounds__(256) void k_step4b(
    const float* __restrict__ CAND, const float* __restrict__ rbuf,
    const float* __restrict__ abuf, const float* __restrict__ cbuf,
    const float* __restrict__ gbuf, const float* __restrict__ valid,
    const float* __restrict__ css, const float* __restrict__ dss,
    const float* __restrict__ xss,
    float* __restrict__ D, float* __restrict__ out, int t) {
  __shared__ float red[256];
  int row = blockIdx.x;                     // (b, m)
  int b = row >> 9, m = row & (M_-1);
  int c = threadIdx.x;
  size_t di = (size_t)row*C_ + c;
  float g  = gbuf[b];
  float rv = rbuf[b*M_+m];
  float SL = dss[b*C_+c], SC = css[b*C_+c], X = xss[b*C_+c];
  float Dold = D[di];
  float cand = CAND[di];
  float dl = Dold + DLR_*rv*abuf[b*C_+c];
  float a1 = (1.f-g) / fmaxf(sqrtf(SL), EPS_);
  float a2 = g       / fmaxf(sqrtf(SC), EPS_);
  float wss = a1*a1*SL + a2*a2*SC + 2.f*a1*a2*X;   // ||blend||^2 analytically
  float inv = 1.f / fmaxf(sqrtf(wss), EPS_);
  float dn = (a1*dl + a2*cand) * inv;
  float v = valid[b*T_+t];
  dn = v*dn + (1.f-v)*Dold;
  D[di] = dn;
  red[c] = dn * cbuf[b*C_+c];
  __syncthreads();
  for (int o = 128; o; o >>= 1) { if (c < o) red[c] += red[c+o]; __syncthreads(); }
  if (c == 0) out[((size_t)(b*T_+t))*M_ + m] = red[0];
}

extern "C" void kernel_launch(void* const* d_in, const int* in_sizes, int n_in,
                              void* d_out, int out_size, void* d_ws, size_t ws_size,
                              hipStream_t stream) {
  (void)in_sizes; (void)n_in; (void)out_size; (void)ws_size;
  const float* u     = (const float*)d_in[0];
  const float* valid = (const float*)d_in[1];
  const float* base  = (const float*)d_in[2];
  const float* Wcu   = (const float*)d_in[3];
  const float* Wcc   = (const float*)d_in[4];
  const float* Wcand_u = (const float*)d_in[5];
  const float* bcu   = (const float*)d_in[6];
  const float* Wcand_c = (const float*)d_in[7];
  const float* bcc   = (const float*)d_in[8];
  const float* Wg    = (const float*)d_in[9];
  const float* bg    = (const float*)d_in[10];
  const float* lnug  = (const float*)d_in[11];
  const float* lnub  = (const float*)d_in[12];
  const float* lncg  = (const float*)d_in[13];
  const float* lncb  = (const float*)d_in[14];
  float* out = (float*)d_out;

  char* ws = (char*)d_ws;
  size_t off = 0;
  auto alloc = [&](size_t bytes) -> char* {
    char* p = ws + off; off = (off + bytes + 255) & ~(size_t)255; return p;
  };
  unsigned short* wu_bf = (unsigned short*)alloc((size_t)MC_*M_*2);  // 128 MB
  unsigned short* wc_bf = (unsigned short*)alloc((size_t)MC_*C_*2);  //  64 MB (L2-resident)
  unsigned short* u_bf  = (unsigned short*)alloc((size_t)B_*T_*M_*2);
  float* P    = (float*)alloc((size_t)B_*T_*MC_*sizeof(float));      // 134 MB
  float* D    = (float*)alloc((size_t)B_*MC_*sizeof(float));         //   4 MB
  float* CAND = (float*)alloc((size_t)B_*MC_*sizeof(float));         //   4 MB
  float* rbuf = (float*)alloc(B_*M_*sizeof(float));
  float* abuf = (float*)alloc(B_*C_*sizeof(float));
  float* cbuf = (float*)alloc(B_*C_*sizeof(float));
  float* css  = (float*)alloc(B_*C_*sizeof(float));
  float* dss  = (float*)alloc(B_*C_*sizeof(float));
  float* xss  = (float*)alloc(B_*C_*sizeof(float));
  float* errsq= (float*)alloc(B_*sizeof(float));
  float* gbuf = (float*)alloc(B_*sizeof(float));
  int*   tki  = (int*)alloc(B_*TOPK_*sizeof(int));
  float* tkv  = (float*)alloc(B_*TOPK_*sizeof(float));

  hipMemsetAsync(cbuf, 0, B_*C_*sizeof(float), stream);   // c_prev = 0

  // one-time weight/input conversion + P precompute (W_cand_u streamed once)
  k_f32_to_bf16<<<32768, 256, 0, stream>>>(Wcand_u, wu_bf, (size_t)MC_*M_/8);
  k_f32_to_bf16<<<16384, 256, 0, stream>>>(Wcand_c, wc_bf, (size_t)MC_*C_/8);
  k_f32_to_bf16<<<64,    256, 0, stream>>>(u,       u_bf,  (size_t)B_*T_*M_/8);
  k_init_D<<<C_, 128, 0, stream>>>(base, D);
  k_precompute_P<<<dim3(2048, 16), 128, 0, stream>>>(u_bf, wu_bf, bcu, bcc, P);

  for (int t = 0; t < T_; ++t) {
    k_step1<<<1, 256, 0, stream>>>(u, valid, Wcu, Wcc, lnug, lnub, lncg, lncb,
                                   cbuf, abuf, tki, tkv, css, dss, xss, errsq, t);
    k_step2<<<16, 256, 0, stream>>>(u, D, tki, tkv, rbuf, errsq, t);
    k_step3<<<1, 256, 0, stream>>>(u, cbuf, Wg, bg, errsq, gbuf, t);
    k_step4a<<<dim3(16, 16), 128, 0, stream>>>(wc_bf, P, cbuf, abuf, D, rbuf,
                                               CAND, css, dss, xss, t);
    k_step4b<<<B_*M_, 256, 0, stream>>>(CAND, rbuf, abuf, cbuf, gbuf, valid,
                                        css, dss, xss, D, out, t);
  }
}
